// PolicyHead_1Trunk_44684839747695
// MI455X (gfx1250) — matrix operand order repaired and checked by reading, not compiled, not run on hardware
//
#include <hip/hip_runtime.h>
#include <hip/hip_bf16.h>

typedef __bf16 bf16_t;
typedef bf16_t v16bf __attribute__((ext_vector_type(16)));
typedef bf16_t v8bf  __attribute__((ext_vector_type(8)));
typedef bf16_t v4bf  __attribute__((ext_vector_type(4)));
typedef float  v8f   __attribute__((ext_vector_type(8)));

#define CCH      256
#define NROWS    (256 * 2048)          // 524288
#define STRIP_M  64
#define NSTRIPS  (NROWS / STRIP_M)     // 8192
#define GBLK     1024
#define ACT      4096
#define NPG      2048
#define BN_EPS   1e-5f

// workspace layout (bytes)
#define WS_WB    0                      // 65536 bf16 swizzled W (stats pass)   131072 B
#define WS_WF    131072                 // 65536 bf16 swizzled folded W         131072 B
#define WS_PART  262144                 // GBLK * 512 floats                    2097152 B
#define WS_A     (262144 + 2097152)     // a[256]                               1024 B
#define WS_BF    (WS_A + 1024)          // folded bias[256]                     1024 B
#define WS_V     (WS_BF + 1024)         // dense v [256*2048] floats            2097152 B

// LDS: [0,131072) swizzled W bf16; [131072,163840) x-tile bf16 (aliased as
// float scratch for reductions after the K loop).
#define LDS_BYTES 163840

// Pre-swizzle lin_w (fp32 [n][k], h = x @ W^T) into the exact WMMA B-fragment
// order so a B fragment is one contiguous 32B load per lane.
// idx = (((kb*16 + tt)*32 + lane)*16 + e) maps to  n = tt*16 + lane%16,
// k = kb*32 + (lane/16)*8 + (e%8) + (e/8)*16   (16-bit B 32x16 layout, CDNA5 ISA 7.12.2)
__global__ void swizzle_w_kernel(const float* __restrict__ lin_w,
                                 const float* __restrict__ aArr,
                                 bf16_t* __restrict__ dst, int use_scale) {
    int idx = blockIdx.x * 256 + threadIdx.x;     // 0..65535
    int e    = idx & 15;
    int lane = (idx >> 4) & 31;
    int tt   = (idx >> 9) & 15;
    int kb   = idx >> 13;
    int n = tt * 16 + (lane & 15);
    int k = kb * 32 + ((lane >> 4) << 3) + (e & 7) + ((e >> 3) << 4);
    float w = lin_w[n * CCH + k];
    if (use_scale) w *= aArr[n];
    dst[idx] = (bf16_t)w;
}

__device__ __forceinline__ v16bf load_a_frag(const bf16_t* arow, int kb, int half) {
    v8bf lo = *(const v8bf*)(arow + kb * 32 + 8 * half);
    v8bf hi = *(const v8bf*)(arow + kb * 32 + 16 + 8 * half);
    v16bf a;
#pragma unroll
    for (int e = 0; e < 8; ++e) { a[e] = lo[e]; a[e + 8] = hi[e]; }
    return a;
}

// MODE 0: stats pass  (accumulate sum/sumsq of h = xW^T + lin_b per channel)
// MODE 1: value pass  (h' = xW'^T + b', relu, dot fin_w, scatter v to vdense)
// Wave tiling: 8 waves = 2 row-groups (32 rows) x 4 col-groups (64 cols).
// Each wave: 2 A frags x 4 B frags -> 8 wmma per kb (each B feeds 2 wmma).
template <int MODE>
__global__ void gemm_kernel(const float* __restrict__ x,
                            const bf16_t* __restrict__ Wsw,
                            const float* __restrict__ bias,
                            const float* __restrict__ finw,
                            const float* __restrict__ finb,
                            const int* __restrict__ batch,
                            float* __restrict__ partials,
                            float* __restrict__ vdense) {
    extern __shared__ char smem[];
    bf16_t* s_w   = (bf16_t*)smem;                 // 131072 B, whole swizzled W
    bf16_t* s_x   = (bf16_t*)(smem + 131072);      // 32768 B, 64x256 bf16 tile
    float*  s_red = (float*)(smem + 131072);       // alias (used outside K loop)

    const int tid  = threadIdx.x;
    const int lane = tid & 31;
    const int wave = tid >> 5;      // 8 waves
    const int rowg = wave >> 2;     // 0..1 : 32-row group within strip
    const int colg = wave & 3;      // 0..3 : 64-col group
    const int half = lane >> 4;
    const int l16  = lane & 15;

    // cooperative load of full swizzled W into LDS (stays for whole kernel)
    {
        const float4* src = (const float4*)Wsw;
        float4* dst = (float4*)s_w;
#pragma unroll
        for (int i = 0; i < 32; ++i) dst[tid + i * 256] = src[tid + i * 256];
    }

    // per-lane column constants: n = colg*64 + t*16 + l16
    float bias_f[4], fw_f[4];
#pragma unroll
    for (int t = 0; t < 4; ++t) {
        int n = colg * 64 + t * 16 + l16;
        bias_f[t] = bias[n];
        fw_f[t] = (MODE == 1) ? finw[n] : 0.f;
    }
    float st_s[4], st_q[4];
#pragma unroll
    for (int t = 0; t < 4; ++t) { st_s[t] = 0.f; st_q[t] = 0.f; }

    // software pipeline: VGPR prefetch of the x strip (16 x float4 per thread)
    float4 R[16];
    {
        const float4* x4 = (const float4*)x + (size_t)blockIdx.x * (STRIP_M * CCH / 4);
#pragma unroll
        for (int i = 0; i < 16; ++i) R[i] = x4[tid + i * 256];
    }

    for (int strip = blockIdx.x; strip < NSTRIPS; strip += GBLK) {
        __syncthreads();           // previous compute done reading s_x
        // convert prefetched strip fp32 -> bf16 into LDS
#pragma unroll
        for (int i = 0; i < 16; ++i) {
            float4 v = R[i];
            v4bf p;
            p[0] = (bf16_t)v.x; p[1] = (bf16_t)v.y;
            p[2] = (bf16_t)v.z; p[3] = (bf16_t)v.w;
            *(v4bf*)(&s_x[(tid + i * 256) * 4]) = p;
        }
        __syncthreads();

        // issue global loads for the NEXT strip; they complete during compute
        {
            int ns = strip + GBLK;
            int ps = (ns < NSTRIPS) ? ns : blockIdx.x;   // clamp (dummy reload)
            const float4* x4 = (const float4*)x + (size_t)ps * (STRIP_M * CCH / 4);
#pragma unroll
            for (int i = 0; i < 16; ++i) R[i] = x4[tid + i * 256];
        }

        v8f acc0[4], acc1[4];
#pragma unroll
        for (int t = 0; t < 4; ++t) { acc0[t] = (v8f)(0.f); acc1[t] = (v8f)(0.f); }

        const bf16_t* arow0 = &s_x[(rowg * 32 + l16) * CCH];
        const bf16_t* arow1 = &s_x[(rowg * 32 + 16 + l16) * CCH];
#pragma unroll
        for (int kb = 0; kb < 8; ++kb) {
            v16bf a0 = load_a_frag(arow0, kb, half);
            v16bf a1 = load_a_frag(arow1, kb, half);
#pragma unroll
            for (int t = 0; t < 4; ++t) {
                int tt = colg * 4 + t;
                v16bf b = *(const v16bf*)(&s_w[((kb * 16 + tt) * 32 + lane) * 16]);
                acc0[t] = __builtin_amdgcn_wmma_f32_16x16x32_bf16(
                    false, a0, false, b, (short)0, acc0[t], false, false);
                acc1[t] = __builtin_amdgcn_wmma_f32_16x16x32_bf16(
                    false, a1, false, b, (short)0, acc1[t], false, false);
            }
        }

        if (MODE == 0) {
            // C layout: vgpr r of lane -> local row (m*16 + half*8 + r), col n
#pragma unroll
            for (int t = 0; t < 4; ++t) {
#pragma unroll
                for (int r = 0; r < 8; ++r) {
                    float h0 = acc0[t][r] + bias_f[t];
                    float h1 = acc1[t][r] + bias_f[t];
                    st_s[t] += h0 + h1;
                    st_q[t] += h0 * h0 + h1 * h1;
                }
            }
        } else {
            float vp0[8], vp1[8];
#pragma unroll
            for (int r = 0; r < 8; ++r) { vp0[r] = 0.f; vp1[r] = 0.f; }
#pragma unroll
            for (int t = 0; t < 4; ++t) {
#pragma unroll
                for (int r = 0; r < 8; ++r) {
                    float h0 = fmaxf(acc0[t][r] + bias_f[t], 0.f);
                    float h1 = fmaxf(acc1[t][r] + bias_f[t], 0.f);
                    vp0[r] += h0 * fw_f[t];
                    vp1[r] += h1 * fw_f[t];
                }
            }
            // reduce over the 16 lanes of this half (columns)
#pragma unroll
            for (int off = 1; off < 16; off <<= 1)
#pragma unroll
                for (int r = 0; r < 8; ++r) {
                    vp0[r] += __shfl_xor(vp0[r], off, 32);
                    vp1[r] += __shfl_xor(vp1[r], off, 32);
                }

            __syncthreads();       // all waves done reading s_x this strip
            float* s_v = s_red;    // 64 floats (aliases s_x)
            if (tid < 64) s_v[tid] = 0.f;
            __syncthreads();
            if (l16 == 0) {
#pragma unroll
                for (int r = 0; r < 8; ++r) {
                    atomicAdd(&s_v[rowg * 32 + half * 8 + r], vp0[r]);
                    atomicAdd(&s_v[rowg * 32 + 16 + half * 8 + r], vp1[r]);
                }
            }
            __syncthreads();
            if (tid < 64) {
                int row = strip * STRIP_M + tid;
                float v = s_v[tid] + finb[0];
                int g = batch[2 * row];
                int aidx = batch[2 * row + 1];   // < NPG by construction
                vdense[(size_t)g * NPG + aidx] = v;  // unique idx: store==add
            }
        }
    }

    if (MODE == 0) {
        // cross-wave per-channel reduction, one partial record per block
        __syncthreads();
        s_red[tid] = 0.f;
        s_red[tid + 256] = 0.f;
        __syncthreads();
#pragma unroll
        for (int t = 0; t < 4; ++t) {
            st_s[t] += __shfl_xor(st_s[t], 16, 32);
            st_q[t] += __shfl_xor(st_q[t], 16, 32);
        }
        if (half == 0) {
#pragma unroll
            for (int t = 0; t < 4; ++t) {
                int n = colg * 64 + t * 16 + l16;
                atomicAdd(&s_red[n], st_s[t]);
                atomicAdd(&s_red[256 + n], st_q[t]);
            }
        }
        __syncthreads();
        float* pp = partials + (size_t)blockIdx.x * 512;
        pp[tid] = s_red[tid];
        pp[tid + 256] = s_red[tid + 256];
    }
}

__global__ void finalize_kernel(const float* __restrict__ partials,
                                const float* __restrict__ gamma,
                                const float* __restrict__ beta,
                                const float* __restrict__ lin_b,
                                float* __restrict__ aArr,
                                float* __restrict__ bfArr) {
    __shared__ float s[512];
    int tid = threadIdx.x;   // 512 threads
    float tot = 0.f;
    for (int g = 0; g < GBLK; ++g) tot += partials[(size_t)g * 512 + tid];
    s[tid] = tot;
    __syncthreads();
    if (tid < 256) {
        const float invN = 1.0f / (float)NROWS;
        float mean = s[tid] * invN;
        float var  = s[tid + 256] * invN - mean * mean;
        float inv  = rsqrtf(var + BN_EPS);
        float a    = gamma[tid] * inv;
        aArr[tid]  = a;
        bfArr[tid] = a * (lin_b[tid] - mean) + beta[tid];
    }
}

// Row has 2048 real values (vdense) + 2048 implicit zeros; handle the zeros
// analytically:  m = max(max_j v, 0);  Z = sum(exp(v-m)) + 2048*exp(-m)
__global__ void logsoftmax_kernel(const float* __restrict__ vdense,
                                  float* __restrict__ out) {
    __shared__ float s[256];
    int g = blockIdx.x, tid = threadIdx.x;
    const float* vr = vdense + (size_t)g * NPG;
    float* row = out + (size_t)g * ACT;
    float v[8];
    float m = 0.f;                      // implicit zeros participate in max
#pragma unroll
    for (int k = 0; k < 8; ++k) {
        v[k] = vr[tid + k * 256];
        m = fmaxf(m, v[k]);
    }
    s[tid] = m;
    __syncthreads();
    for (int off = 128; off > 0; off >>= 1) {
        if (tid < off) s[tid] = fmaxf(s[tid], s[tid + off]);
        __syncthreads();
    }
    m = s[0];
    __syncthreads();
    float sum = 0.f;
#pragma unroll
    for (int k = 0; k < 8; ++k) sum += __expf(v[k] - m);
    s[tid] = sum;
    __syncthreads();
    for (int off = 128; off > 0; off >>= 1) {
        if (tid < off) s[tid] += s[tid + off];
        __syncthreads();
    }
    float lse = m + __logf(s[0] + (float)NPG * __expf(-m));
    float neg = -lse;
#pragma unroll
    for (int k = 0; k < 8; ++k) {
        row[tid + k * 256] = v[k] - lse;          // real entries
        row[NPG + tid + k * 256] = neg;           // implicit-zero entries
    }
}

extern "C" void kernel_launch(void* const* d_in, const int* in_sizes, int n_in,
                              void* d_out, int out_size, void* d_ws, size_t ws_size,
                              hipStream_t stream) {
    (void)in_sizes; (void)n_in; (void)out_size; (void)ws_size;
    const float* x     = (const float*)d_in[0];
    const int*   batch = (const int*)d_in[1];
    const float* lin_w = (const float*)d_in[2];
    const float* lin_b = (const float*)d_in[3];
    const float* gamma = (const float*)d_in[4];
    const float* beta  = (const float*)d_in[5];
    const float* fin_w = (const float*)d_in[6];
    const float* fin_b = (const float*)d_in[7];
    float* out = (float*)d_out;
    char* ws = (char*)d_ws;
    bf16_t* Wb       = (bf16_t*)(ws + WS_WB);
    bf16_t* Wf       = (bf16_t*)(ws + WS_WF);
    float*  partials = (float*)(ws + WS_PART);
    float*  aArr     = (float*)(ws + WS_A);
    float*  bfArr    = (float*)(ws + WS_BF);
    float*  vdense   = (float*)(ws + WS_V);

    // swizzled bf16 weights for the stats GEMM
    swizzle_w_kernel<<<256, 256, 0, stream>>>(lin_w, aArr, Wb, 0);
    // pass 1: GEMM + per-channel sum / sumsq partials
    gemm_kernel<0><<<GBLK, 256, LDS_BYTES, stream>>>(
        x, Wb, lin_b, nullptr, nullptr, nullptr, partials, nullptr);
    // reduce partials -> BN fold (a, b')
    finalize_kernel<<<1, 512, 0, stream>>>(partials, gamma, beta, lin_b, aArr, bfArr);
    // fold BN scale into weights
    swizzle_w_kernel<<<256, 256, 0, stream>>>(lin_w, aArr, Wf, 1);
    // pass 2: folded GEMM + relu + fin_w dot + scatter v (dense) via batch
    gemm_kernel<1><<<GBLK, 256, LDS_BYTES, stream>>>(
        x, Wf, bfArr, fin_w, fin_b, batch, nullptr, vdense);
    // log_softmax with analytic handling of the 2048 implicit zeros per row
    logsoftmax_kernel<<<256, 256, 0, stream>>>(vdense, out);
}
